// EdgeBiasTransformerLayer_28355374088842
// MI455X (gfx1250) — matrix operand order repaired and checked by reading, not compiled, not run on hardware
//
#include <hip/hip_runtime.h>
#include <hip/hip_bf16.h>
#include <math.h>

// ---------------------------------------------------------------------------
// EdgeBiasTransformerLayer for MI455X (gfx1250, wave32, WMMA bf16 16x16x32)
// ---------------------------------------------------------------------------

typedef __attribute__((ext_vector_type(16))) __bf16 v16bf;
typedef __attribute__((ext_vector_type(8)))  float  v8f;

#define D_MODEL 768
#define NTOK    4096      // B*N
#define SEQ     1024
#define NHEAD   12
#define DK      64
#define DFF     3072
#define NBATCH  4

// ---------------- bf16 helpers (storage = unsigned short) -------------------
static __device__ __forceinline__ unsigned short f2bf(float f) {
    unsigned int u = __float_as_uint(f);
    u += 0x7FFFu + ((u >> 16) & 1u);           // round-to-nearest-even
    return (unsigned short)(u >> 16);
}
static __device__ __forceinline__ float bf2f(unsigned short h) {
    return __uint_as_float(((unsigned int)h) << 16);
}

static __device__ __forceinline__ v8f vzero8() {
    v8f z;
#pragma unroll
    for (int i = 0; i < 8; ++i) z[i] = 0.0f;
    return z;
}

// A fragment: 16x32 (MxK) bf16, row-major A with leading dim lda.
// lane<16  holds row m0+l, K = k0+0..7  and k0+16..23
// lane>=16 holds row m0+l, K = k0+8..15 and k0+24..31
static __device__ __forceinline__ v16bf load_a_frag(const unsigned short* A, int lda,
                                                    int m0, int k0, int lane) {
    int hl = lane >> 4, l = lane & 15;
    const unsigned short* p = A + (size_t)(m0 + l) * lda + k0 + hl * 8;
    union { uint4 u[2]; v16bf v; } t;
    t.u[0] = *(const uint4*)(p);
    t.u[1] = *(const uint4*)(p + 16);
    return t.v;
}

// B fragment: 32x16 (KxN) bf16 where B = W^T and W is [Nout x K] row-major.
// lane<16  holds col n0+l, K = k0+0..7  and k0+16..23
// lane>=16 holds col n0+l, K = k0+8..15 and k0+24..31
static __device__ __forceinline__ v16bf load_b_frag(const unsigned short* W, int ldw,
                                                    int n0, int k0, int lane) {
    int hl = lane >> 4, l = lane & 15;
    const unsigned short* p = W + (size_t)(n0 + l) * ldw + k0 + hl * 8;
    union { uint4 u[2]; v16bf v; } t;
    t.u[0] = *(const uint4*)(p);
    t.u[1] = *(const uint4*)(p + 16);
    return t.v;
}

static __device__ __forceinline__ v8f wmma_bf16(v16bf a, v16bf b, v8f c) {
    return __builtin_amdgcn_wmma_f32_16x16x32_bf16(false, a, false, b,
                                                   (short)0, c, false, false);
}

// ---------------- elementwise fp32 -> bf16 convert --------------------------
__global__ __launch_bounds__(256) void cvt_bf16(const float* __restrict__ src,
                                                unsigned short* __restrict__ dst, int n) {
    int i = blockIdx.x * 256 + threadIdx.x;
    if (i < n) dst[i] = f2bf(src[i]);
}

// ---------------- LayerNorm (fp32 in, bf16 out), D = 768 --------------------
__global__ __launch_bounds__(256) void ln_bf16(const float* __restrict__ x,
                                               const float* __restrict__ g,
                                               const float* __restrict__ bta,
                                               unsigned short* __restrict__ out) {
    int row = blockIdx.x, t = threadIdx.x;
    const float* xr = x + (size_t)row * D_MODEL;
    float v0 = xr[t], v1 = xr[t + 256], v2 = xr[t + 512];
    __shared__ float r1[256], r2[256];
    r1[t] = v0 + v1 + v2;
    r2[t] = v0 * v0 + v1 * v1 + v2 * v2;
    __syncthreads();
    for (int off = 128; off > 0; off >>= 1) {
        if (t < off) { r1[t] += r1[t + off]; r2[t] += r2[t + off]; }
        __syncthreads();
    }
    float mu  = r1[0] * (1.0f / 768.0f);
    float var = r2[0] * (1.0f / 768.0f) - mu * mu;
    float inv = rsqrtf(var + 1e-5f);
    unsigned short* o = out + (size_t)row * D_MODEL;
    o[t]       = f2bf((v0 - mu) * inv * g[t]       + bta[t]);
    o[t + 256] = f2bf((v1 - mu) * inv * g[t + 256] + bta[t + 256]);
    o[t + 512] = f2bf((v2 - mu) * inv * g[t + 512] + bta[t + 512]);
}

// ---------------- generic WMMA GEMM: C = act(A * W^T + bias) (+resid) -------
// A: [M x K] bf16 row-major, W: [Nout x K] bf16 row-major.
// 8 waves/block; wave w computes rows m0..m0+15 x cols n0..n0+63 (4 WMMA tiles)
template <int ACT, int RESID>
__global__ __launch_bounds__(256) void gemm_bf16(const unsigned short* __restrict__ A,
                                                 const unsigned short* __restrict__ W,
                                                 const float* __restrict__ bias,
                                                 const float* __restrict__ resid,
                                                 float* __restrict__ out_f,
                                                 unsigned short* __restrict__ out_b,
                                                 int Nout, int K) {
    int w = threadIdx.x >> 5, lane = threadIdx.x & 31;
    int hl = lane >> 4, l = lane & 15;
    int m0 = blockIdx.x * 128 + w * 16;
    int n0 = blockIdx.y * 64;
    v8f c[4];
#pragma unroll
    for (int j = 0; j < 4; ++j) c[j] = vzero8();

    for (int k0 = 0; k0 < K; k0 += 32) {
        v16bf a = load_a_frag(A, K, m0, k0, lane);
#pragma unroll
        for (int j = 0; j < 4; ++j) {
            v16bf bw = load_b_frag(W, K, n0 + j * 16, k0, lane);
            c[j] = wmma_bf16(a, bw, c[j]);
        }
    }
#pragma unroll
    for (int j = 0; j < 4; ++j) {
        int col = n0 + j * 16 + l;
        float bv = bias[col];
#pragma unroll
        for (int r = 0; r < 8; ++r) {
            int row = m0 + hl * 8 + r;
            float v = c[j][r] + bv;
            if (ACT == 1) v = 0.5f * v * (1.0f + erff(v * 0.70710678118f)); // exact GELU
            size_t o = (size_t)row * Nout + col;
            if (RESID) v += resid[o];
            if (out_f) out_f[o] = v;
            if (out_b) out_b[o] = f2bf(v);
        }
    }
}

// ---------------- V transpose: [tok][H*64] -> [B,H,64][N] -------------------
__global__ __launch_bounds__(256) void transpose_v(const unsigned short* __restrict__ v,
                                                   unsigned short* __restrict__ vT) {
    int o = blockIdx.x * 256 + threadIdx.x;           // 4*12*64*1024 outputs
    int n  = o & 1023;
    int t  = o >> 10;
    int d  = t & 63;
    int bh = t >> 6;
    int h  = bh % NHEAD;
    int b  = bh / NHEAD;
    vT[o] = v[(size_t)(b * SEQ + n) * D_MODEL + h * DK + d];
}

// ---------------- edge bias (memory bound: reads 256MB once) ----------------
// bias[b,h,n,m] = edge_feats[b,n,m,:] . We[h] + be[h], masked -> -1e30, bf16
__global__ __launch_bounds__(256) void edge_bias_k(const float* __restrict__ ef,
                                                   const unsigned char* __restrict__ pad,
                                                   const unsigned char* __restrict__ adj,
                                                   const float* __restrict__ We,
                                                   const float* __restrict__ be,
                                                   unsigned short* __restrict__ bias) {
    __shared__ float sW[NHEAD * 16];
    __shared__ float sB[NHEAD];
    if (threadIdx.x < NHEAD * 16) sW[threadIdx.x] = We[threadIdx.x];
    if (threadIdx.x < NHEAD)      sB[threadIdx.x] = be[threadIdx.x];
    __syncthreads();

    size_t tid = (size_t)blockIdx.x * 256 + threadIdx.x;   // over B*N*N
    int m = (int)(tid & 1023);
    int n = (int)((tid >> 10) & 1023);
    int b = (int)(tid >> 20);

    const float4* e4 = (const float4*)(ef + tid * 16);
    float4 e0 = e4[0], e1 = e4[1], e2 = e4[2], e3 = e4[3];
    float ev[16] = {e0.x, e0.y, e0.z, e0.w, e1.x, e1.y, e1.z, e1.w,
                    e2.x, e2.y, e2.z, e2.w, e3.x, e3.y, e3.z, e3.w};
    bool masked = (adj[tid] == 0) || (pad[((size_t)b << 10) + m] != 0);

#pragma unroll
    for (int h = 0; h < NHEAD; ++h) {
        float acc = sB[h];
#pragma unroll
        for (int i = 0; i < 16; ++i) acc = fmaf(ev[i], sW[h * 16 + i], acc);
        if (masked) acc = -1e30f;
        bias[((((size_t)b * NHEAD + h) * SEQ + n) << 10) + m] = f2bf(acc);
    }
}

// ---------------- flash attention: 1 wave per (b, h, 16-row q tile) ---------
__global__ __launch_bounds__(32) void flash_attn(const unsigned short* __restrict__ q,
                                                 const unsigned short* __restrict__ k,
                                                 const unsigned short* __restrict__ vT,
                                                 const unsigned short* __restrict__ bias,
                                                 unsigned short* __restrict__ out) {
    int lane = threadIdx.x & 31;
    int hl = lane >> 4, l = lane & 15;
    int bid = blockIdx.x;
    int qt = bid & 63;
    int hb = bid >> 6;
    int h  = hb % NHEAD;
    int b  = hb / NHEAD;
    int tokq = b * SEQ + qt * 16;

    // Q fragments for K-dim 0..31 / 32..63; fold 1/sqrt(64) into Q (exact in bf16)
    v16bf a0 = load_a_frag(q, D_MODEL, tokq, h * DK,      lane);
    v16bf a1 = load_a_frag(q, D_MODEL, tokq, h * DK + 32, lane);
#pragma unroll
    for (int i = 0; i < 16; ++i) {
        a0[i] = (__bf16)((float)a0[i] * 0.125f);
        a1[i] = (__bf16)((float)a1[i] * 0.125f);
    }

    float mst[8], lst[8];
    v8f o[4];
#pragma unroll
    for (int r = 0; r < 8; ++r) { mst[r] = -3.0e38f; lst[r] = 0.0f; }
#pragma unroll
    for (int j = 0; j < 4; ++j) o[j] = vzero8();

    __shared__ float pt[16][32];                 // P tile for C->A relayout
    const unsigned short* biasb =
        bias + (((size_t)(b * NHEAD + h) * SEQ + qt * 16) << 10);

    for (int kb = 0; kb < SEQ; kb += 32) {
        // ---- S = Q K^T / sqrt(dk): two 16x16 sub-tiles, K-dim split 32+32
        v8f s0, s1;
        {
            v16bf bk;
            bk = load_b_frag(k, D_MODEL, b * SEQ + kb,      h * DK,      lane);
            s0 = wmma_bf16(a0, bk, vzero8());
            bk = load_b_frag(k, D_MODEL, b * SEQ + kb,      h * DK + 32, lane);
            s0 = wmma_bf16(a1, bk, s0);
            bk = load_b_frag(k, D_MODEL, b * SEQ + kb + 16, h * DK,      lane);
            s1 = wmma_bf16(a0, bk, vzero8());
            bk = load_b_frag(k, D_MODEL, b * SEQ + kb + 16, h * DK + 32, lane);
            s1 = wmma_bf16(a1, bk, s1);
        }
        // ---- add precomputed (masked) edge bias
#pragma unroll
        for (int r = 0; r < 8; ++r) {
            const unsigned short* bp = biasb + ((size_t)(hl * 8 + r) << 10) + kb + l;
            s0[r] += bf2f(bp[0]);
            s1[r] += bf2f(bp[16]);
        }
        // ---- online softmax: row r lives across 16 lanes of this half-wave
        float p0[8], p1[8];
#pragma unroll
        for (int r = 0; r < 8; ++r) {
            float tm = fmaxf(s0[r], s1[r]);
            tm = fmaxf(tm, __shfl_xor(tm, 1, 32));
            tm = fmaxf(tm, __shfl_xor(tm, 2, 32));
            tm = fmaxf(tm, __shfl_xor(tm, 4, 32));
            tm = fmaxf(tm, __shfl_xor(tm, 8, 32));
            float mn = fmaxf(mst[r], tm);
            bool dead = (mn < -1e29f);           // fully-masked row -> nan_to_num(0)
            float es = __expf(mst[r] - mn);
            p0[r] = dead ? 0.0f : __expf(s0[r] - mn);
            p1[r] = dead ? 0.0f : __expf(s1[r] - mn);
            float ls = p0[r] + p1[r];
            ls += __shfl_xor(ls, 1, 32);
            ls += __shfl_xor(ls, 2, 32);
            ls += __shfl_xor(ls, 4, 32);
            ls += __shfl_xor(ls, 8, 32);
            lst[r] = lst[r] * es + ls;
            mst[r] = mn;
            o[0][r] *= es; o[1][r] *= es; o[2][r] *= es; o[3][r] *= es;
        }
        // ---- relayout P (C layout) -> A layout via LDS
        __syncthreads();
#pragma unroll
        for (int r = 0; r < 8; ++r) {
            pt[hl * 8 + r][l]      = p0[r];
            pt[hl * 8 + r][16 + l] = p1[r];
        }
        __syncthreads();
        v16bf pa;
#pragma unroll
        for (int e = 0; e < 8; ++e) {
            pa[e]     = (__bf16)pt[l][hl * 8 + e];
            pa[8 + e] = (__bf16)pt[l][16 + hl * 8 + e];
        }
        // ---- O += P * V   (V^T layout gives contiguous B fragments)
#pragma unroll
        for (int j = 0; j < 4; ++j) {
            v16bf bv = load_b_frag(vT, SEQ, (b * NHEAD + h) * DK + j * 16, kb, lane);
            o[j] = wmma_bf16(pa, bv, o[j]);
        }
    }
    // ---- normalize + store
#pragma unroll
    for (int r = 0; r < 8; ++r) {
        float inv = (lst[r] > 0.0f) ? (1.0f / lst[r]) : 0.0f;
#pragma unroll
        for (int j = 0; j < 4; ++j) {
            out[(size_t)(tokq + hl * 8 + r) * D_MODEL + h * DK + j * 16 + l] =
                f2bf(o[j][r] * inv);
        }
    }
}

// ---------------------------------------------------------------------------
extern "C" void kernel_launch(void* const* d_in, const int* in_sizes, int n_in,
                              void* d_out, int out_size, void* d_ws, size_t ws_size,
                              hipStream_t stream) {
    (void)in_sizes; (void)n_in; (void)out_size; (void)ws_size;
    const float* x   = (const float*)d_in[0];
    const float* ef  = (const float*)d_in[1];
    const unsigned char* pad = (const unsigned char*)d_in[2];
    const unsigned char* adj = (const unsigned char*)d_in[3];
    const float* Wq = (const float*)d_in[4];  const float* bq  = (const float*)d_in[5];
    const float* Wk = (const float*)d_in[6];  const float* bk  = (const float*)d_in[7];
    const float* Wv = (const float*)d_in[8];  const float* bv  = (const float*)d_in[9];
    const float* Wo = (const float*)d_in[10]; const float* bo  = (const float*)d_in[11];
    const float* We = (const float*)d_in[12]; const float* be  = (const float*)d_in[13];
    const float* g1 = (const float*)d_in[14]; const float* bt1 = (const float*)d_in[15];
    const float* W1 = (const float*)d_in[16]; const float* fb1 = (const float*)d_in[17];
    const float* W2 = (const float*)d_in[18]; const float* fb2 = (const float*)d_in[19];
    const float* g2 = (const float*)d_in[20]; const float* bt2 = (const float*)d_in[21];

    // ---- workspace carve-up (~188 MB, 256B aligned) ----
    char* wp = (char*)d_ws;
    auto bump = [&](size_t bytes) -> void* {
        void* p = (void*)wp;
        wp += (bytes + 255) & ~(size_t)255;
        return p;
    };
    unsigned short* Wq_b = (unsigned short*)bump((size_t)768 * 768 * 2);
    unsigned short* Wk_b = (unsigned short*)bump((size_t)768 * 768 * 2);
    unsigned short* Wv_b = (unsigned short*)bump((size_t)768 * 768 * 2);
    unsigned short* Wo_b = (unsigned short*)bump((size_t)768 * 768 * 2);
    unsigned short* W1_b = (unsigned short*)bump((size_t)3072 * 768 * 2);
    unsigned short* W2_b = (unsigned short*)bump((size_t)768 * 3072 * 2);
    unsigned short* h_b  = (unsigned short*)bump((size_t)NTOK * 768 * 2);
    unsigned short* q_b  = (unsigned short*)bump((size_t)NTOK * 768 * 2);
    unsigned short* k_b  = (unsigned short*)bump((size_t)NTOK * 768 * 2);
    unsigned short* v_b  = (unsigned short*)bump((size_t)NTOK * 768 * 2);
    unsigned short* vT_b = (unsigned short*)bump((size_t)NTOK * 768 * 2);
    unsigned short* ao_b = (unsigned short*)bump((size_t)NTOK * 768 * 2);
    unsigned short* h2_b = (unsigned short*)bump((size_t)NTOK * 768 * 2);
    unsigned short* ff_b = (unsigned short*)bump((size_t)NTOK * 3072 * 2);
    float*          x1   = (float*)bump((size_t)NTOK * 768 * 4);
    unsigned short* bias_b = (unsigned short*)bump((size_t)NBATCH * NHEAD * SEQ * SEQ * 2);

    // ---- 1. weights -> bf16 ----
    cvt_bf16<<<2304, 256, 0, stream>>>(Wq, Wq_b, 768 * 768);
    cvt_bf16<<<2304, 256, 0, stream>>>(Wk, Wk_b, 768 * 768);
    cvt_bf16<<<2304, 256, 0, stream>>>(Wv, Wv_b, 768 * 768);
    cvt_bf16<<<2304, 256, 0, stream>>>(Wo, Wo_b, 768 * 768);
    cvt_bf16<<<9216, 256, 0, stream>>>(W1, W1_b, 3072 * 768);
    cvt_bf16<<<9216, 256, 0, stream>>>(W2, W2_b, 768 * 3072);

    // ---- 2. LN1 ----
    ln_bf16<<<NTOK, 256, 0, stream>>>(x, g1, bt1, h_b);

    // ---- 3. Q/K/V projections (WMMA) ----
    gemm_bf16<0, 0><<<dim3(32, 12), 256, 0, stream>>>(h_b, Wq_b, bq, nullptr, nullptr, q_b, 768, 768);
    gemm_bf16<0, 0><<<dim3(32, 12), 256, 0, stream>>>(h_b, Wk_b, bk, nullptr, nullptr, k_b, 768, 768);
    gemm_bf16<0, 0><<<dim3(32, 12), 256, 0, stream>>>(h_b, Wv_b, bv, nullptr, nullptr, v_b, 768, 768);
    transpose_v<<<12288, 256, 0, stream>>>(v_b, vT_b);

    // ---- 4. edge bias + masks -> bf16 [B,H,N,N] (reads edge_feats once) ----
    edge_bias_k<<<16384, 256, 0, stream>>>(ef, pad, adj, We, be, bias_b);

    // ---- 5. flash attention (WMMA QK^T and PV) ----
    flash_attn<<<NBATCH * NHEAD * (SEQ / 16), 32, 0, stream>>>(q_b, k_b, vT_b, bias_b, ao_b);

    // ---- 6. output projection + residual -> x1 (f32) ----
    gemm_bf16<0, 1><<<dim3(32, 12), 256, 0, stream>>>(ao_b, Wo_b, bo, x, x1, nullptr, 768, 768);

    // ---- 7. LN2 ----
    ln_bf16<<<NTOK, 256, 0, stream>>>(x1, g2, bt2, h2_b);

    // ---- 8. FFN: GELU(h2 W1^T + b1) W2^T + b2 + x1 -> d_out ----
    gemm_bf16<1, 0><<<dim3(32, 48), 256, 0, stream>>>(h2_b, W1_b, fb1, nullptr, nullptr, ff_b, 3072, 768);
    gemm_bf16<0, 1><<<dim3(32, 12), 256, 0, stream>>>(ff_b, W2_b, fb2, x1, (float*)d_out, nullptr, 768, 3072);
}
